// LSTM_25941602468487
// MI455X (gfx1250) — matrix operand order repaired and checked by reading, not compiled, not run on hardware
//
#include <hip/hip_runtime.h>

// ---------------------------------------------------------------------------
// Fused 2-layer LSTM + FC head for MI455X (gfx1250, wave32, WMMA).
// One workgroup = 16 batch rows (one WMMA M-tile) x full T recurrence.
// Grid = 512/16 = 32 workgroups, 256 threads (8 waves) each.
//   waves 0..6 : gate GEMM N-tiles (4H=100 -> 7 tiles of 16), WMMA f16->f32
//   waves 0..3 : FC N-tiles (51 -> 4 tiles of 16)
// Weights live in VGPRs for the whole loop; h/c/z/emb/indices live in LDS.
// Gate nonlinearity uses v_exp_f32 + v_rcp_f32 (no IEEE-div expansion).
// ---------------------------------------------------------------------------

typedef _Float16 f16;
typedef _Float16 v16h __attribute__((ext_vector_type(16)));
typedef _Float16 h8v  __attribute__((ext_vector_type(8)));
typedef float    v8f  __attribute__((ext_vector_type(8)));

union V16 { v16h v; h8v h[2]; };

#define BB    512
#define TT    2048
#define GPAD  112          // 4H=100 padded to 7 N-tiles
#define ZST   116          // LDS row stride for z (floats), avoids bank conflicts

// workspace byte offsets (prep kernel output, all 16B aligned)
#define O_W0X 0                          // [112][64] f16  Wih0 padded (N,K)
#define O_W0H (O_W0X + GPAD*64*2)        // [112][32] f16  Whh0
#define O_W1X (O_W0H + GPAD*32*2)        // [112][32] f16  Wih1
#define O_W1H (O_W1X + GPAD*32*2)        // [112][32] f16  Whh1
#define O_FCW (O_W1H + GPAD*32*2)        // [ 64][32] f16  fcW
#define O_EMB (O_FCW + 64*32*2)          // [ 52][64] f16  embedding table
#define O_BS0 (O_EMB + 52*64*2)          // [112] f32      bih0+bhh0
#define O_BS1 (O_BS0 + GPAD*4)           // [112] f32      bih1+bhh1
#define O_FCB (O_BS1 + GPAD*4)           // [ 64] f32      fcb

// sigmoid via v_exp_f32 + v_rcp_f32 (hardware rcp, ~1 ulp; plenty for LSTM)
__device__ __forceinline__ float sigm_f(float x) {
  return __builtin_amdgcn_rcpf(1.f + __expf(-x));
}
__device__ __forceinline__ float tanh_f(float x) {
  return 2.f * __builtin_amdgcn_rcpf(1.f + __expf(-2.f * x)) - 1.f;
}

#define WMMA(A, Bm, C) __builtin_amdgcn_wmma_f32_16x16x32_f16(false, (A), false, (Bm), (short)0, (C), false, false)

// ------------------------- weight pre-pack kernel --------------------------
__global__ __launch_bounds__(256) void lstm_prep(
    const float* __restrict__ emb,
    const float* __restrict__ Wih0, const float* __restrict__ Whh0,
    const float* __restrict__ bih0, const float* __restrict__ bhh0,
    const float* __restrict__ Wih1, const float* __restrict__ Whh1,
    const float* __restrict__ bih1, const float* __restrict__ bhh1,
    const float* __restrict__ fcW,  const float* __restrict__ fcb,
    char* __restrict__ ws)
{
  const int tid = threadIdx.x;
  f16* w0x = (f16*)(ws + O_W0X);
  f16* w0h = (f16*)(ws + O_W0H);
  f16* w1x = (f16*)(ws + O_W1X);
  f16* w1h = (f16*)(ws + O_W1H);
  f16* fcw = (f16*)(ws + O_FCW);
  f16* emp = (f16*)(ws + O_EMB);
  float* b0 = (float*)(ws + O_BS0);
  float* b1 = (float*)(ws + O_BS1);
  float* fb = (float*)(ws + O_FCB);

  for (int i = tid; i < GPAD*64; i += 256) {            // Wih0: [100][50] -> [112][64]
    int n = i >> 6, k = i & 63;
    w0x[i] = (f16)((n < 100 && k < 50) ? Wih0[n*50 + k] : 0.f);
  }
  for (int i = tid; i < GPAD*32; i += 256) {            // Whh0 / Wih1 / Whh1: [100][25] -> [112][32]
    int n = i >> 5, k = i & 31;
    bool ok = (n < 100 && k < 25);
    w0h[i] = (f16)(ok ? Whh0[n*25 + k] : 0.f);
    w1x[i] = (f16)(ok ? Wih1[n*25 + k] : 0.f);
    w1h[i] = (f16)(ok ? Whh1[n*25 + k] : 0.f);
  }
  for (int i = tid; i < 64*32; i += 256) {              // fcW: [51][25] -> [64][32]
    int n = i >> 5, k = i & 31;
    fcw[i] = (f16)((n < 51 && k < 25) ? fcW[n*25 + k] : 0.f);
  }
  for (int i = tid; i < 52*64; i += 256) {              // emb: [51][50] -> [52][64]
    int r = i >> 6, k = i & 63;
    emp[i] = (f16)((r < 51 && k < 50) ? emb[r*50 + k] : 0.f);
  }
  for (int i = tid; i < GPAD; i += 256) {
    b0[i] = (i < 100) ? (bih0[i] + bhh0[i]) : 0.f;
    b1[i] = (i < 100) ? (bih1[i] + bhh1[i]) : 0.f;
  }
  for (int i = tid; i < 64; i += 256) fb[i] = (i < 51) ? fcb[i] : 0.f;
}

// ----------------------------- main LSTM kernel ----------------------------
__global__ __launch_bounds__(256) void lstm_main(
    const int* __restrict__ x, const char* __restrict__ ws, float* __restrict__ out)
{
  const f16* w0x = (const f16*)(ws + O_W0X);
  const f16* w0h = (const f16*)(ws + O_W0H);
  const f16* w1x = (const f16*)(ws + O_W1X);
  const f16* w1h = (const f16*)(ws + O_W1H);
  const f16* fcw = (const f16*)(ws + O_FCW);
  const f16* embp = (const f16*)(ws + O_EMB);
  const float* bs0 = (const float*)(ws + O_BS0);
  const float* bs1 = (const float*)(ws + O_BS1);
  const float* fcb = (const float*)(ws + O_FCB);

  __shared__ alignas(16) f16 s_emb[52*64];        // f16 embedding table
  __shared__ unsigned char   s_xidx[TT*16];       // [t][row] note indices (u8)
  __shared__ alignas(16) float s_z0[16*ZST];      // layer0 gates
  __shared__ alignas(16) float s_z1[16*ZST];      // layer1 gates
  __shared__ alignas(16) f16 s_h0[16*32];         // h0 (f16, zero-padded cols 25..31)
  __shared__ alignas(16) f16 s_h1[16*32];         // h1 (f16)
  __shared__ float s_c0[16*32];
  __shared__ float s_c1[16*32];
  __shared__ float s_h0f[16*32];
  __shared__ float s_h1f[16*32];

  const int tid  = threadIdx.x;
  const int wave = tid >> 5;
  const int lane = tid & 31;
  const int nl   = lane & 15;   // row (A) / column (B,C,D) index within tile
  const int kh   = lane >> 4;   // K-half / M-half selector
  const int b0   = blockIdx.x * 16;

  // ---- one-time LDS setup ----
  for (int i = tid; i < 52*64; i += 256) s_emb[i] = embp[i];
  for (int i = tid; i < 16*TT; i += 256) {                    // coalesced over t
    int r = i >> 11, t = i & (TT - 1);
    s_xidx[t*16 + r] = (unsigned char)x[(size_t)(b0 + r)*TT + t];
  }
  for (int i = tid; i < 16*32; i += 256) {
    s_h0[i] = (f16)0.f; s_h1[i] = (f16)0.f;
    s_c0[i] = 0.f; s_c1[i] = 0.f; s_h0f[i] = 0.f; s_h1f[i] = 0.f;
  }

  // ---- loop-invariant weight B-tiles into VGPRs ----
  // B-tile (16-bit, K=32): lane holds column nl, K = kh*8..kh*8+7 and 16+kh*8..23+kh*8 (table order)
  V16 W0Xlo = {}, W0Xhi = {}, W0H = {}, W1X = {}, W1H = {}, FCWr = {};
  float bias0 = 0.f, bias1 = 0.f, biasF = 0.f;
  if (wave < 7) {
    const int col = wave*16 + nl;
    const h8v* p;
    p = (const h8v*)(w0x + col*64 + kh*8);       W0Xlo.h[0] = p[0]; W0Xlo.h[1] = p[2];
    p = (const h8v*)(w0x + col*64 + 32 + kh*8);  W0Xhi.h[0] = p[0]; W0Xhi.h[1] = p[2];
    p = (const h8v*)(w0h + col*32 + kh*8);       W0H.h[0]   = p[0]; W0H.h[1]   = p[2];
    p = (const h8v*)(w1x + col*32 + kh*8);       W1X.h[0]   = p[0]; W1X.h[1]   = p[2];
    p = (const h8v*)(w1h + col*32 + kh*8);       W1H.h[0]   = p[0]; W1H.h[1]   = p[2];
    bias0 = bs0[col]; bias1 = bs1[col];
  }
  if (wave < 4) {
    const int col = wave*16 + nl;
    const h8v* p = (const h8v*)(fcw + col*32 + kh*8);
    FCWr.h[0] = p[0]; FCWr.h[1] = p[2];
    biasF = fcb[col];
  }
  __syncthreads();

  // A-tile (16-bit, K=32): lane holds row nl; VGPR0-3 = K kh*8..kh*8+7,
  // VGPR4-7 = K 16+kh*8..23+kh*8  -> two ds_load_b128 at +0/+32B (row base + kh*16B)
  for (int t = 0; t < TT; ++t) {
    // ---------- layer 0 gates: z0 = emb(x_t)@W0x^T + h0@W0h^T + b ----------
    if (wave < 7) {
      const int idx = (int)s_xidx[t*16 + nl];
      V16 Ae_lo, Ae_hi, Ah;
      {
        const h8v* ea = (const h8v*)(s_emb + idx*64 + kh*8);
        Ae_lo.h[0] = ea[0]; Ae_lo.h[1] = ea[2];     // K 0..31
        Ae_hi.h[0] = ea[4]; Ae_hi.h[1] = ea[6];     // K 32..63
      }
      {
        const h8v* ha = (const h8v*)(s_h0 + nl*32 + kh*8);
        Ah.h[0] = ha[0]; Ah.h[1] = ha[2];
      }
      v8f C = { bias0, bias0, bias0, bias0, bias0, bias0, bias0, bias0 };
      C = WMMA(Ae_lo.v, W0Xlo.v, C);
      C = WMMA(Ae_hi.v, W0Xhi.v, C);
      C = WMMA(Ah.v,    W0H.v,   C);
      const int col = wave*16 + nl;
      #pragma unroll
      for (int v = 0; v < 8; ++v) s_z0[(v + 8*kh)*ZST + col] = C[v];
    }
    __syncthreads();
    // ---------- layer 0 nonlinearity ----------
    #pragma unroll
    for (int k = 0; k < 2; ++k) {
      int e = tid + k*256;
      if (e < 400) {
        int m = e / 25, j = e - m*25;
        const float* zr = s_z0 + m*ZST;
        float iv = zr[j], fv = zr[25+j], gv = zr[50+j], ov = zr[75+j];
        float c = sigm_f(fv) * s_c0[m*32+j] + sigm_f(iv) * tanh_f(gv);
        float h = sigm_f(ov) * tanh_f(c);
        s_c0[m*32+j] = c; s_h0f[m*32+j] = h; s_h0[m*32+j] = (f16)h;
      }
    }
    __syncthreads();
    // ---------- layer 1 gates: z1 = h0@W1x^T + h1@W1h^T + b ----------
    if (wave < 7) {
      V16 Ax, Ah;
      const h8v* xa = (const h8v*)(s_h0 + nl*32 + kh*8);
      Ax.h[0] = xa[0]; Ax.h[1] = xa[2];
      const h8v* ha = (const h8v*)(s_h1 + nl*32 + kh*8);
      Ah.h[0] = ha[0]; Ah.h[1] = ha[2];
      v8f C = { bias1, bias1, bias1, bias1, bias1, bias1, bias1, bias1 };
      C = WMMA(Ax.v, W1X.v, C);
      C = WMMA(Ah.v, W1H.v, C);
      const int col = wave*16 + nl;
      #pragma unroll
      for (int v = 0; v < 8; ++v) s_z1[(v + 8*kh)*ZST + col] = C[v];
    }
    __syncthreads();
    // ---------- layer 1 nonlinearity ----------
    #pragma unroll
    for (int k = 0; k < 2; ++k) {
      int e = tid + k*256;
      if (e < 400) {
        int m = e / 25, j = e - m*25;
        const float* zr = s_z1 + m*ZST;
        float iv = zr[j], fv = zr[25+j], gv = zr[50+j], ov = zr[75+j];
        float c = sigm_f(fv) * s_c1[m*32+j] + sigm_f(iv) * tanh_f(gv);
        float h = sigm_f(ov) * tanh_f(c);
        s_c1[m*32+j] = c; s_h1f[m*32+j] = h; s_h1[m*32+j] = (f16)h;
      }
    }
    __syncthreads();
    // ---------- FC head: logits_t = h1 @ fcW^T + fcb ----------
    if (wave < 4) {
      V16 Ay;
      const h8v* ya = (const h8v*)(s_h1 + nl*32 + kh*8);
      Ay.h[0] = ya[0]; Ay.h[1] = ya[2];
      v8f C = { biasF, biasF, biasF, biasF, biasF, biasF, biasF, biasF };
      C = WMMA(Ay.v, FCWr.v, C);
      const int col = wave*16 + nl;
      if (col < 51) {
        #pragma unroll
        for (int v = 0; v < 8; ++v) {
          const int m = v + 8*kh;
          out[((size_t)(b0 + m)*TT + t)*51 + col] = C[v];
        }
      }
    }
    // no barrier needed: next step's writers are >=2 barriers downstream of readers
  }

  // ---- final h_n / c_n : [2,B,25] each, appended after logits ----
  __syncthreads();
  const size_t logN = (size_t)BB * TT * 51;
  #pragma unroll
  for (int k = 0; k < 2; ++k) {
    int e = tid + k*256;
    if (e < 400) {
      int m = e / 25, j = e - m*25;
      size_t o = (size_t)(b0 + m)*25 + j;
      out[logN + o]                      = s_h0f[m*32+j];
      out[logN + (size_t)BB*25 + o]      = s_h1f[m*32+j];
      out[logN + (size_t)2*BB*25 + o]    = s_c0[m*32+j];
      out[logN + (size_t)3*BB*25 + o]    = s_c1[m*32+j];
    }
  }
}

// ------------------------------- launcher ----------------------------------
extern "C" void kernel_launch(void* const* d_in, const int* in_sizes, int n_in,
                              void* d_out, int out_size, void* d_ws, size_t ws_size,
                              hipStream_t stream) {
  (void)in_sizes; (void)n_in; (void)out_size; (void)ws_size;
  const int*   x    = (const int*)  d_in[0];
  const float* emb  = (const float*)d_in[1];
  const float* Wih0 = (const float*)d_in[2];
  const float* Whh0 = (const float*)d_in[3];
  const float* bih0 = (const float*)d_in[4];
  const float* bhh0 = (const float*)d_in[5];
  const float* Wih1 = (const float*)d_in[6];
  const float* Whh1 = (const float*)d_in[7];
  const float* bih1 = (const float*)d_in[8];
  const float* bhh1 = (const float*)d_in[9];
  const float* fcW  = (const float*)d_in[10];
  const float* fcb  = (const float*)d_in[11];
  char* ws = (char*)d_ws;

  lstm_prep<<<1, 256, 0, stream>>>(emb, Wih0, Whh0, bih0, bhh0,
                                   Wih1, Whh1, bih1, bhh1, fcW, fcb, ws);
  lstm_main<<<32, 256, 0, stream>>>(x, ws, (float*)d_out);
}
